// vGINMolHeadEncoder_28278064677193
// MI455X (gfx1250) — compile-verified, matrix operand order checked by reading, not run on hardware
//
#include <hip/hip_runtime.h>
#include <hip/hip_bf16.h>
#include <stdint.h>

#define NN 100000     // nodes
#define NE 1600000    // edges
#define DD 128        // emb dim
#define LLAYER 5
#define GG 512        // graphs
#define AV 119        // atom vocab
#define BV 5          // bond vocab
#define BNEPS 1e-5f

typedef __attribute__((ext_vector_type(16))) __bf16 v16bf;
typedef __attribute__((ext_vector_type(8)))  float  v8f;
typedef __attribute__((ext_vector_type(4)))  float  f4raw;   // trivial 16B vector

union FragBF { v16bf v; f4raw q[2]; };

static __device__ __forceinline__ __bf16 f2bf(float f) {
    union { float f; uint32_t u; } in; in.f = f;
    uint32_t r = in.u + 0x7FFFu + ((in.u >> 16) & 1u);   // round-to-nearest-even
    union { uint16_t s; __bf16 b; } out; out.s = (uint16_t)(r >> 16);
    return out.b;
}

// ---------------------------------------------------------------------------
// Atom encoder: h[n][c] = sum_{i<9} atom_emb[i][x[n][i]][c]
// ---------------------------------------------------------------------------
__global__ void atom_encode_kernel(const int* __restrict__ x,
                                   const float* __restrict__ emb,
                                   float* __restrict__ h) {
    long long idx = (long long)blockIdx.x * blockDim.x + threadIdx.x;
    if (idx >= (long long)NN * (DD / 4)) return;
    int n = (int)(idx >> 5);
    int c = ((int)idx & 31) << 2;
    f4raw s; s.x = 0.f; s.y = 0.f; s.z = 0.f; s.w = 0.f;
#pragma unroll
    for (int i = 0; i < 9; ++i) {
        int v = x[n * 9 + i];
        f4raw t = *(const f4raw*)(emb + ((size_t)i * AV + v) * DD + c);
        s.x += t.x; s.y += t.y; s.z += t.z; s.w += t.w;
    }
    *(f4raw*)(h + (size_t)n * DD + c) = s;
}

// vn[g][c] = vn_emb[c]
__global__ void vn_init_kernel(const float* __restrict__ vn_emb, float* __restrict__ vn) {
    int idx = blockIdx.x * blockDim.x + threadIdx.x;
    if (idx >= GG * DD) return;
    vn[idx] = vn_emb[idx & (DD - 1)];
}

// hB[n] = hA[n] + vn[batch[n]]
__global__ void add_vn_kernel(const float* __restrict__ hA, const float* __restrict__ vn,
                              const int* __restrict__ batch, float* __restrict__ hB) {
    long long idx = (long long)blockIdx.x * blockDim.x + threadIdx.x;
    if (idx >= (long long)NN * (DD / 4)) return;
    int n = (int)(idx >> 5);
    int c = ((int)idx & 31) << 2;
    int g = batch[n];
    f4raw a = *(const f4raw*)(hA + (size_t)n * DD + c);
    f4raw v = *(const f4raw*)(vn + (size_t)g * DD + c);
    a.x += v.x; a.y += v.y; a.z += v.z; a.w += v.w;
    *(f4raw*)(hB + (size_t)n * DD + c) = a;
}

// ---------------------------------------------------------------------------
// Edge phase: msg = relu(h[src] + bond0[a0] + bond1[a1] + bond2[a2]);
// atomicAdd into agg[dst].  agg + h both fit in 192MB L2.
// ---------------------------------------------------------------------------
__global__ void edge_scatter_kernel(const float* __restrict__ h,
                                    const int* __restrict__ ei,
                                    const int* __restrict__ ea,
                                    const float* __restrict__ bond,  // [3][BV][DD] (this layer)
                                    float* __restrict__ agg) {
    long long idx = (long long)blockIdx.x * blockDim.x + threadIdx.x;
    if (idx >= (long long)NE * (DD / 4)) return;
    int e = (int)(idx >> 5);
    int c = ((int)idx & 31) << 2;
    int src = ei[e];
    int dst = ei[NE + e];
    int a0 = ea[e * 3 + 0], a1 = ea[e * 3 + 1], a2 = ea[e * 3 + 2];
    f4raw hv = *(const f4raw*)(h + (size_t)src * DD + c);
    f4raw b0 = *(const f4raw*)(bond + ((size_t)0 * BV + a0) * DD + c);
    f4raw b1 = *(const f4raw*)(bond + ((size_t)1 * BV + a1) * DD + c);
    f4raw b2 = *(const f4raw*)(bond + ((size_t)2 * BV + a2) * DD + c);
    float m0 = fmaxf(hv.x + b0.x + b1.x + b2.x, 0.f);
    float m1 = fmaxf(hv.y + b0.y + b1.y + b2.y, 0.f);
    float m2 = fmaxf(hv.z + b0.z + b1.z + b2.z, 0.f);
    float m3 = fmaxf(hv.w + b0.w + b1.w + b2.w, 0.f);
    float* ap = agg + (size_t)dst * DD + c;
    atomicAdd(ap + 0, m0);
    atomicAdd(ap + 1, m1);
    atomicAdd(ap + 2, m2);
    atomicAdd(ap + 3, m3);
}

// z = (1+eps)*h + agg, quantized to bf16 (GEMM-A input)
__global__ void combine_z_kernel(const float* __restrict__ hp, const float* __restrict__ agg,
                                 const float* __restrict__ epsArr, int layer,
                                 __bf16* __restrict__ out) {
    long long idx = (long long)blockIdx.x * blockDim.x + threadIdx.x;
    if (idx >= (long long)NN * DD) return;
    float ev = 1.0f + epsArr[layer];
    out[idx] = f2bf(ev * hp[idx] + agg[idx]);
}

// pooled[batch[n]] += h[n]
__global__ void pool_scatter_kernel(const float* __restrict__ h, const int* __restrict__ batch,
                                    float* __restrict__ pooled) {
    long long idx = (long long)blockIdx.x * blockDim.x + threadIdx.x;
    if (idx >= (long long)NN * (DD / 4)) return;
    int n = (int)(idx >> 5);
    int c = ((int)idx & 31) << 2;
    int g = batch[n];
    f4raw v = *(const f4raw*)(h + (size_t)n * DD + c);
    float* pp = pooled + (size_t)g * DD + c;
    atomicAdd(pp + 0, v.x);
    atomicAdd(pp + 1, v.y);
    atomicAdd(pp + 2, v.z);
    atomicAdd(pp + 3, v.w);
}

// bf16(pooled + vn)  (vn-MLP GEMM-A input)
__global__ void vn_gemm_in_kernel(const float* __restrict__ pooled, const float* __restrict__ vn,
                                  __bf16* __restrict__ out) {
    int idx = blockIdx.x * blockDim.x + threadIdx.x;
    if (idx >= GG * DD) return;
    out[idx] = f2bf(pooled[idx] + vn[idx]);
}

// ---------------------------------------------------------------------------
// Weight packing: fp32 W[K x C] -> bf16 fragments matching the ISA B-matrix
// layout for v_wmma_f32_16x16x32_bf16.  Per (ntile, ktile): 32 lanes x 16 bf16.
// ---------------------------------------------------------------------------
__global__ void pack_w_kernel(const float* __restrict__ W, __bf16* __restrict__ out,
                              int K, int C) {
    int idx = blockIdx.x * blockDim.x + threadIdx.x;
    if (idx >= K * C) return;
    int k = idx / C, n = idx % C;
    int kt = k >> 5, kin = k & 31;
    int sub16 = kin >> 4;                 // selects VGPR group 0-3 vs 4-7
    int w16 = kin & 15;
    int lane = (n & 15) + ((w16 >> 3) << 4);
    int e = (sub16 << 3) + (w16 & 7);
    int nt = n >> 4;
    int nKt = K >> 5;
    size_t off = (((size_t)(nt * nKt + kt)) << 9) + ((size_t)lane << 4) + e;
    out[off] = f2bf(W[idx]);
}

// ---------------------------------------------------------------------------
// WMMA GEMM: C[M x C] = A[M x K](bf16, row-major) @ Wpacked + bias
// One wave -> 16(M) x 64(N) strip.  Fully unrolled K-loop with an explicit
// double-buffered pipeline: A/B fragments for step kt+1 are loaded before the
// 4 WMMAs of step kt issue, so VMEM overlaps the matrix pipe.
// amdgpu_waves_per_eu(4) lifts the occupancy-driven VGPR cap (~120 regs live).
// Requires M % 16 == 0 (100000 = 6250*16, 512 = 32*16), C % 64 == 0.
// ---------------------------------------------------------------------------
template <int KVAL>
__global__ void __launch_bounds__(256)
__attribute__((amdgpu_waves_per_eu(4)))
wmma_gemm_bf16_kernel(const __bf16* __restrict__ A, const __bf16* __restrict__ Bp,
                      const float* __restrict__ bias, float* __restrict__ Cout,
                      int M, int C) {
    constexpr int nKt = KVAL / 32;
    const int lane = threadIdx.x & 31;
    const int wave = threadIdx.x >> 5;
    const int nQuads = C >> 6;
    const int w = blockIdx.x * 8 + wave;
    if (w >= (M >> 4) * nQuads) return;
    const int mt = w / nQuads;
    const int nq = w % nQuads;

    const int row = (mt << 4) + (lane & 15);
    const int kHalf = (lane >> 4) << 3;     // 0 or 8

    // loop-invariant base pointers; all loads below use immediate offsets
    const __bf16* aBase = A + ((size_t)row << (KVAL == 128 ? 7 : 8)) + kHalf;
    const __bf16* bBase = Bp + (((size_t)(nq * 4) * nKt) << 9) + ((size_t)lane << 4);

    v8f acc[4];
#pragma unroll
    for (int j = 0; j < 4; ++j)
#pragma unroll
        for (int i = 0; i < 8; ++i) acc[j][i] = 0.0f;

    FragBF aC, aN;
    FragBF bC[4], bN[4];

    // prologue: stage 0 loads
    aC.q[0] = *(const f4raw*)(aBase);
    aC.q[1] = *(const f4raw*)(aBase + 16);
#pragma unroll
    for (int j = 0; j < 4; ++j) {
        const __bf16* bp = bBase + ((size_t)(j * nKt) << 9);
        bC[j].q[0] = *(const f4raw*)(bp);
        bC[j].q[1] = *(const f4raw*)(bp + 8);
    }

#pragma unroll
    for (int kt = 0; kt < nKt; ++kt) {
        if (kt + 1 < nKt) {
            aN.q[0] = *(const f4raw*)(aBase + ((kt + 1) << 5));
            aN.q[1] = *(const f4raw*)(aBase + ((kt + 1) << 5) + 16);
#pragma unroll
            for (int j = 0; j < 4; ++j) {
                const __bf16* bp = bBase + ((size_t)(j * nKt + kt + 1) << 9);
                bN[j].q[0] = *(const f4raw*)(bp);
                bN[j].q[1] = *(const f4raw*)(bp + 8);
            }
        }
#pragma unroll
        for (int j = 0; j < 4; ++j)
            acc[j] = __builtin_amdgcn_wmma_f32_16x16x32_bf16(
                false, aC.v, false, bC[j].v, (short)0, acc[j], false, false);
        if (kt + 1 < nKt) {
            aC = aN;
#pragma unroll
            for (int j = 0; j < 4; ++j) bC[j] = bN[j];
        }
    }

    const int rbase = (mt << 4) + ((lane >> 4) << 3);
#pragma unroll
    for (int j = 0; j < 4; ++j) {
        const int col = (((nq << 2) + j) << 4) + (lane & 15);
        const float bv = bias[col];
#pragma unroll
        for (int i = 0; i < 8; ++i)
            Cout[(size_t)(rbase + i) * C + col] = acc[j][i] + bv;
    }
}

// ---------------------------------------------------------------------------
// BatchNorm: column sums / sum-of-squares, then normalize (+optional ReLU,
// optional fp32 and/or bf16 outputs).
// ---------------------------------------------------------------------------
__global__ void col_stats_kernel(const float* __restrict__ X, float* __restrict__ stats,
                                 int M, int C) {
    int c = threadIdx.x;                      // blockDim.x == C
    float s = 0.f, sq = 0.f;
    for (int r = blockIdx.x; r < M; r += gridDim.x) {
        float v = X[(size_t)r * C + c];
        s += v; sq += v * v;
    }
    atomicAdd(&stats[c], s);
    atomicAdd(&stats[C + c], sq);
}

__global__ void bn_apply_kernel(const float* __restrict__ X, const float* __restrict__ stats,
                                const float* __restrict__ g, const float* __restrict__ b,
                                float* __restrict__ outF, __bf16* __restrict__ outBf,
                                int M, int C, int doRelu) {
    long long idx = (long long)blockIdx.x * blockDim.x + threadIdx.x;
    if (idx >= (long long)M * C) return;
    int c = (int)(idx % C);
    float invM = 1.0f / (float)M;
    float mu = stats[c] * invM;
    float var = stats[C + c] * invM - mu * mu;
    float y = (X[idx] - mu) * rsqrtf(var + BNEPS) * g[c] + b[c];
    if (doRelu) y = fmaxf(y, 0.f);
    if (outF) outF[idx] = y;
    if (outBf) outBf[idx] = f2bf(y);
}

// ---------------------------------------------------------------------------
// Host side
// ---------------------------------------------------------------------------
static inline int nblocks(long long total, int tpb) { return (int)((total + tpb - 1) / tpb); }

extern "C" void kernel_launch(void* const* d_in, const int* in_sizes, int n_in,
                              void* d_out, int out_size, void* d_ws, size_t ws_size,
                              hipStream_t stream) {
    (void)in_sizes; (void)n_in; (void)out_size; (void)ws_size;

    const int*   x        = (const int*)d_in[0];
    const int*   ei       = (const int*)d_in[1];
    const int*   ea       = (const int*)d_in[2];
    const int*   batch    = (const int*)d_in[3];
    const float* atom_emb = (const float*)d_in[4];
    const float* bond_emb = (const float*)d_in[5];
    const float* epsArr   = (const float*)d_in[6];
    const float* w1       = (const float*)d_in[7];
    const float* b1       = (const float*)d_in[8];
    const float* bn1_g    = (const float*)d_in[9];
    const float* bn1_b    = (const float*)d_in[10];
    const float* w2       = (const float*)d_in[11];
    const float* b2       = (const float*)d_in[12];
    const float* obn_g    = (const float*)d_in[13];
    const float* obn_b    = (const float*)d_in[14];
    const float* vn_emb   = (const float*)d_in[15];
    const float* vw1      = (const float*)d_in[16];
    const float* vb1      = (const float*)d_in[17];
    const float* vbn1_g   = (const float*)d_in[18];
    const float* vbn1_b   = (const float*)d_in[19];
    const float* vw2      = (const float*)d_in[20];
    const float* vb2      = (const float*)d_in[21];
    const float* vbn2_g   = (const float*)d_in[22];
    const float* vbn2_b   = (const float*)d_in[23];

    float* out = (float*)d_out;

    // workspace carve-out
    size_t off = 0;
    auto take = [&](size_t bytes) -> void* {
        off = (off + 255) & ~(size_t)255;
        void* p = (char*)d_ws + off;
        off += bytes;
        return p;
    };
    float*  hA     = (float*)take((size_t)NN * DD * 4);
    float*  hB     = (float*)take((size_t)NN * DD * 4);
    float*  agg    = (float*)take((size_t)NN * DD * 4);
    float*  t1     = (float*)take((size_t)NN * 2 * DD * 4);
    __bf16* abf    = (__bf16*)take((size_t)NN * 2 * DD * 2);
    __bf16* wpack  = (__bf16*)take((size_t)12 * 32768 * 2);
    float*  stats  = (float*)take(512 * 4);
    float*  vn     = (float*)take((size_t)GG * DD * 4);
    float*  pooled = (float*)take((size_t)GG * DD * 4);
    float*  vt     = (float*)take((size_t)GG * 2 * DD * 4);
    __bf16* vabf   = (__bf16*)take((size_t)GG * 2 * DD * 2);

    const int TPB = 256;
    const size_t WKC = 32768;   // elements per packed weight matrix (128*256)

    // ---- pack all weights to bf16 WMMA-B fragments (once per launch) ----
    for (int l = 0; l < LLAYER; ++l) {
        pack_w_kernel<<<nblocks(WKC, TPB), TPB, 0, stream>>>(
            w1 + (size_t)l * WKC, wpack + (size_t)l * WKC, 128, 256);
        pack_w_kernel<<<nblocks(WKC, TPB), TPB, 0, stream>>>(
            w2 + (size_t)l * WKC, wpack + (size_t)(LLAYER + l) * WKC, 256, 128);
    }
    pack_w_kernel<<<nblocks(WKC, TPB), TPB, 0, stream>>>(vw1, wpack + 10 * WKC, 128, 256);
    pack_w_kernel<<<nblocks(WKC, TPB), TPB, 0, stream>>>(vw2, wpack + 11 * WKC, 256, 128);

    // ---- atom encoder + virtual-node init ----
    atom_encode_kernel<<<nblocks((long long)NN * 32, TPB), TPB, 0, stream>>>(x, atom_emb, hA);
    vn_init_kernel<<<nblocks(GG * DD, TPB), TPB, 0, stream>>>(vn_emb, vn);

    const int gemmBlk1 = nblocks((long long)(NN / 16) * 4, 8);  // C=256
    const int gemmBlk2 = nblocks((long long)(NN / 16) * 2, 8);  // C=128
    const int vgemm1   = nblocks((long long)(GG / 16) * 4, 8);
    const int vgemm2   = nblocks((long long)(GG / 16) * 2, 8);

    for (int l = 0; l < LLAYER; ++l) {
        const float* hin = hA;
        if (l > 0) {
            add_vn_kernel<<<nblocks((long long)NN * 32, TPB), TPB, 0, stream>>>(hA, vn, batch, hB);
            hin = hB;
        }

        // edge message + scatter-add
        hipMemsetAsync(agg, 0, (size_t)NN * DD * 4, stream);
        edge_scatter_kernel<<<nblocks((long long)NE * 32, TPB), TPB, 0, stream>>>(
            hin, ei, ea, bond_emb + (size_t)l * 3 * BV * DD, agg);

        // z = (1+eps)h + agg -> bf16
        combine_z_kernel<<<nblocks((long long)NN * DD, TPB), TPB, 0, stream>>>(
            hin, agg, epsArr, l, abf);

        // GEMM1: [N,128] @ [128,256] + b1
        wmma_gemm_bf16_kernel<128><<<gemmBlk1, TPB, 0, stream>>>(
            abf, wpack + (size_t)l * WKC, b1 + (size_t)l * 256, t1, NN, 256);

        // BN1 + ReLU -> bf16
        hipMemsetAsync(stats, 0, 512 * 4, stream);
        col_stats_kernel<<<1024, 256, 0, stream>>>(t1, stats, NN, 256);
        bn_apply_kernel<<<nblocks((long long)NN * 256, TPB), TPB, 0, stream>>>(
            t1, stats, bn1_g + (size_t)l * 256, bn1_b + (size_t)l * 256,
            (float*)nullptr, abf, NN, 256, 1);

        // GEMM2: [N,256] @ [256,128] + b2
        wmma_gemm_bf16_kernel<256><<<gemmBlk2, TPB, 0, stream>>>(
            abf, wpack + (size_t)(LLAYER + l) * WKC, b2 + (size_t)l * 128, hB, NN, 128);

        // outer BN (+ReLU except last layer); last layer writes d_out
        hipMemsetAsync(stats, 0, 512 * 4, stream);
        col_stats_kernel<<<1024, 128, 0, stream>>>(hB, stats, NN, 128);
        if (l == LLAYER - 1) {
            bn_apply_kernel<<<nblocks((long long)NN * DD, TPB), TPB, 0, stream>>>(
                hB, stats, obn_g + (size_t)l * 128, obn_b + (size_t)l * 128,
                out, (__bf16*)nullptr, NN, 128, 0);
        } else {
            bn_apply_kernel<<<nblocks((long long)NN * DD, TPB), TPB, 0, stream>>>(
                hB, stats, obn_g + (size_t)l * 128, obn_b + (size_t)l * 128,
                hA, (__bf16*)nullptr, NN, 128, 1);
        }

        // virtual-node update after layers 1..3
        if (l >= 1 && l <= 3) {
            hipMemsetAsync(pooled, 0, (size_t)GG * DD * 4, stream);
            pool_scatter_kernel<<<nblocks((long long)NN * 32, TPB), TPB, 0, stream>>>(hA, batch, pooled);
            vn_gemm_in_kernel<<<nblocks(GG * DD, TPB), TPB, 0, stream>>>(pooled, vn, vabf);

            wmma_gemm_bf16_kernel<128><<<vgemm1, TPB, 0, stream>>>(
                vabf, wpack + 10 * WKC, vb1, vt, GG, 256);
            hipMemsetAsync(stats, 0, 512 * 4, stream);
            col_stats_kernel<<<64, 256, 0, stream>>>(vt, stats, GG, 256);
            bn_apply_kernel<<<nblocks((long long)GG * 256, TPB), TPB, 0, stream>>>(
                vt, stats, vbn1_g, vbn1_b, (float*)nullptr, vabf, GG, 256, 1);

            wmma_gemm_bf16_kernel<256><<<vgemm2, TPB, 0, stream>>>(
                vabf, wpack + 11 * WKC, vb2, pooled, GG, 128);
            hipMemsetAsync(stats, 0, 512 * 4, stream);
            col_stats_kernel<<<64, 128, 0, stream>>>(pooled, stats, GG, 128);
            bn_apply_kernel<<<nblocks((long long)GG * DD, TPB), TPB, 0, stream>>>(
                pooled, stats, vbn2_g, vbn2_b, vn, (__bf16*)nullptr, GG, 128, 1);
        }
    }
}
